// MarketAwareGraph_88021059764652
// MI455X (gfx1250) — matrix operand order repaired and checked by reading, not compiled, not run on hardware
//
#include <hip/hip_runtime.h>
#include <hip/hip_bf16.h>
#include <math.h>

typedef __attribute__((ext_vector_type(16))) __bf16 v16bf;
typedef __attribute__((ext_vector_type(8)))  __bf16 v8bf;
typedef __attribute__((ext_vector_type(8)))  float  v8f;

#define B_  8
#define L_  20
#define N_  2048
#define F_  64
#define NN_ (N_ * N_)

__device__ __forceinline__ v16bf mk16(v8bf lo, v8bf hi) {
    return __builtin_shufflevector(lo, hi, 0, 1, 2, 3, 4, 5, 6, 7,
                                           8, 9, 10, 11, 12, 13, 14, 15);
}
__device__ __forceinline__ v8bf ld8(const __bf16* p) {
    return *(const v8bf*)p;   // 16-byte aligned vector load -> global_load_b128
}

// ---------------------------------------------------------------------------
// K0: m2[b,l,f] = mean over n of x_seq[b,l,n,f]
// ---------------------------------------------------------------------------
__global__ void k_mean(const float* __restrict__ x, float* __restrict__ m2) {
    int bl = blockIdx.x;              // 0 .. B*L-1
    int f  = threadIdx.x;             // 0 .. 63
    const float* base = x + (size_t)bl * N_ * F_ + f;
    float s = 0.f;
    for (int n = 0; n < N_; ++n) s += base[(size_t)n * F_];
    m2[bl * F_ + f] = s * (1.0f / (float)N_);
}

// ---------------------------------------------------------------------------
// K1: tiny conv head -> logits -> keep_ratio -> k per batch (one block/batch)
// ---------------------------------------------------------------------------
__global__ void k_feats(const float* __restrict__ m2,
                        const float* __restrict__ w1, const float* __restrict__ b1,
                        const float* __restrict__ w2, const float* __restrict__ b2,
                        const float* __restrict__ w3, const float* __restrict__ b3,
                        const float* __restrict__ w4, const float* __restrict__ b4,
                        const float* __restrict__ w_out, const float* __restrict__ b_out,
                        int* __restrict__ kvals) {
    __shared__ float sfeat[32];
    int b   = blockIdx.x;
    int tid = threadIdx.x;
    if (tid < 32) sfeat[tid] = 0.f;
    __syncthreads();
    const float* img = m2 + b * L_ * F_;   // (H=L, W=F), C=1
    const int HW = L_ * F_;
    for (int idx = tid; idx < 4 * 8 * HW; idx += blockDim.x) {
        int conv = idx / (8 * HW);
        int rem  = idx % (8 * HW);
        int ch   = rem / HW;
        int hw   = rem % HW;
        int h = hw / F_, w = hw % F_;
        float v;
        if (conv == 0) {
            v = img[h * F_ + w] * w1[ch] + b1[ch];
        } else if (conv == 1) {
            float s = 0.f;
            for (int dw = -1; dw <= 1; ++dw) {
                int ww = w + dw;
                if (ww >= 0 && ww < F_) s += img[h * F_ + ww] * w2[ch * 3 + (dw + 1)];
            }
            v = s + b2[ch];
        } else if (conv == 2) {
            float s = 0.f;
            for (int dh = -1; dh <= 1; ++dh) {
                int hh = h + dh;
                if (hh >= 0 && hh < L_) s += img[hh * F_ + w] * w3[ch * 3 + (dh + 1)];
            }
            v = s + b3[ch];
        } else {
            float s = 0.f;
            for (int dh = -1; dh <= 1; ++dh)
                for (int dw = -1; dw <= 1; ++dw) {
                    int hh = h + dh, ww = w + dw;
                    if (hh >= 0 && hh < L_ && ww >= 0 && ww < F_)
                        s += img[hh * F_ + ww] * w4[ch * 9 + (dh + 1) * 3 + (dw + 1)];
                }
            v = s + b4[ch];
        }
        float g = 0.5f * v * (1.0f + erff(v * 0.7071067811865475f));  // exact GELU
        atomicAdd(&sfeat[conv * 8 + ch], g * (1.0f / (float)HW));
    }
    __syncthreads();
    if (tid == 0) {
        float logit = b_out[0];
        for (int i = 0; i < 32; ++i) logit += sfeat[i] * w_out[i];
        float sig = 1.0f / (1.0f + expf(-logit));
        float kr  = fminf(fmaxf(0.2f * sig, 0.02f), 1.0f);
        int   kk  = (int)ceilf(kr * (float)NN_);
        if (kk < 1)   kk = 1;
        if (kk > NN_) kk = NN_;
        kvals[b] = kk;
    }
}

// ---------------------------------------------------------------------------
// K2: row-normalize x_last -> bf16, written BOTH row-major (N,F) and
//     transposed (F,N) per batch.  One wave per (b,n) row.
// ---------------------------------------------------------------------------
__global__ void k_normalize(const float* __restrict__ x,
                            __bf16* __restrict__ xn,    // (B,N,F)
                            __bf16* __restrict__ xt) {  // (B,F,N)
    int row  = blockIdx.x;            // 0 .. B*N-1
    int b    = row / N_, n = row % N_;
    int lane = threadIdx.x;
    const float* src = x + (((size_t)b * L_ + (L_ - 1)) * N_ + n) * F_;
    float v0 = src[lane], v1 = src[lane + 32];
    float ss = v0 * v0 + v1 * v1;
    #pragma unroll
    for (int off = 16; off > 0; off >>= 1) ss += __shfl_xor(ss, off, 32);
    float inv = 1.0f / fmaxf(sqrtf(ss), 1e-12f);
    __bf16 e0 = (__bf16)(v0 * inv), e1 = (__bf16)(v1 * inv);
    __bf16* dn = xn + (size_t)row * F_;
    dn[lane]      = e0;
    dn[lane + 32] = e1;
    __bf16* dt = xt + (size_t)b * F_ * N_;
    dt[(size_t)lane * N_ + n]        = e0;
    dt[(size_t)(lane + 32) * N_ + n] = e1;
}

// ---------------------------------------------------------------------------
// K3: prior_d (f32, row-major) -> bf16 TRANSPOSED (Pt[col][row]).
//     LDS-tiled 32x32 transpose, coalesced both directions.
// ---------------------------------------------------------------------------
__global__ void k_transpose_bf16(const float* __restrict__ src, __bf16* __restrict__ dst) {
    __shared__ __bf16 tile[32][33];
    int bx = blockIdx.x * 32;   // column block of src
    int by = blockIdx.y * 32;   // row block of src
    int tx = threadIdx.x, ty = threadIdx.y;     // block (32,8)
    #pragma unroll
    for (int j = 0; j < 32; j += 8)
        tile[ty + j][tx] = (__bf16)src[(size_t)(by + ty + j) * N_ + bx + tx];
    __syncthreads();
    #pragma unroll
    for (int j = 0; j < 32; j += 8)
        dst[(size_t)(bx + ty + j) * N_ + by + tx] = tile[tx][ty + j];
}

// ---------------------------------------------------------------------------
// K4: T = Xn^T @ P per batch (64 x 2048, K=2048), stored transposed:
//     Tt[b][col*64 + f].  One wave per 16-column tile; all 4 row tiles
//     (full M=64) with 4 accumulators sharing each B fragment.
//     Software-pipelined: next iteration's fragments are in flight while the
//     current WMMAs issue (wraparound prefetch keeps control flow uniform).
// ---------------------------------------------------------------------------
__global__ void k_gemm1(const __bf16* __restrict__ xt,   // (B,F,N)  A rows contiguous in K
                        const __bf16* __restrict__ pt,   // (N,N)    P column-major
                        __bf16* __restrict__ tt) {       // (B,N,F)  Tt
    int colTile = blockIdx.x;        // 0..127
    int b       = blockIdx.y;
    int lane = threadIdx.x & 31;
    int m    = lane & 15;
    int hi   = lane >> 4;
    const __bf16* Xt = xt + (size_t)b * F_ * N_;
    int col = colTile * 16 + m;
    const __bf16* bcol = pt + (size_t)col * N_;          // column `col` of P

    const __bf16* arow[4];
    #pragma unroll
    for (int rt = 0; rt < 4; ++rt) arow[rt] = Xt + (size_t)(rt * 16 + m) * N_;

    v8f acc[4] = {{}, {}, {}, {}};
    // prologue: fragments for n0 = 0
    v16bf bcur = mk16(ld8(bcol + 8 * hi), ld8(bcol + 16 + 8 * hi));
    v16bf acur[4];
    #pragma unroll
    for (int rt = 0; rt < 4; ++rt)
        acur[rt] = mk16(ld8(arow[rt] + 8 * hi), ld8(arow[rt] + 16 + 8 * hi));

    for (int n0 = 0; n0 < N_; n0 += 32) {
        int nn = (n0 + 32) & (N_ - 1);        // wraparound prefetch (uniform)
        v16bf bnext = mk16(ld8(bcol + nn + 8 * hi),
                           ld8(bcol + nn + 16 + 8 * hi));
        v16bf anext[4];
        #pragma unroll
        for (int rt = 0; rt < 4; ++rt)
            anext[rt] = mk16(ld8(arow[rt] + nn + 8 * hi),
                             ld8(arow[rt] + nn + 16 + 8 * hi));
        #pragma unroll
        for (int rt = 0; rt < 4; ++rt)
            acc[rt] = __builtin_amdgcn_wmma_f32_16x16x32_bf16(
                false, acur[rt], false, bcur, (short)0, acc[rt], false, false);
        bcur = bnext;
        #pragma unroll
        for (int rt = 0; rt < 4; ++rt) acur[rt] = anext[rt];
    }
    // Store Tt: lane's 8 accum values per row-tile are CONTIGUOUS in f -> b128 store
    __bf16* Tb = tt + (size_t)b * N_ * F_;
    #pragma unroll
    for (int rt = 0; rt < 4; ++rt) {
        v8bf out;
        #pragma unroll
        for (int r = 0; r < 8; ++r) out[r] = (__bf16)acc[rt][r];
        *(v8bf*)(Tb + (size_t)col * F_ + rt * 16 + 8 * hi) = out;
    }
}

// ---------------------------------------------------------------------------
// K5: a = Xn @ T per batch (2048 x 2048, K=64).  Each wave covers a 16x64
//     strip: one A fragment pair reused across 4 column tiles -> 8 WMMAs.
// ---------------------------------------------------------------------------
__global__ void k_gemm2(const __bf16* __restrict__ xn,   // (B,N,F)
                        const __bf16* __restrict__ tt,   // (B,N,F) = Tt
                        float* __restrict__ aout) {      // (B,N,N)
    int wave    = threadIdx.x >> 5;
    int lane    = threadIdx.x & 31;
    int ct0     = (blockIdx.x * 8 + wave) * 4;   // first of 4 col tiles (0..124)
    int rowTile = blockIdx.y;                    // 0..127
    int b       = blockIdx.z;
    int m  = lane & 15;
    int hi = lane >> 4;
    const __bf16* arow = xn + (size_t)b * N_ * F_ + (size_t)(rowTile * 16 + m) * F_;
    const __bf16* tb   = tt + (size_t)b * N_ * F_;

    // A fragments for k0 = 0 and k0 = 32 (reused across all 4 column tiles)
    v16bf a0 = mk16(ld8(arow + 8 * hi),      ld8(arow + 16 + 8 * hi));
    v16bf a1 = mk16(ld8(arow + 32 + 8 * hi), ld8(arow + 48 + 8 * hi));

    v8f acc[4] = {{}, {}, {}, {}};
    #pragma unroll
    for (int c = 0; c < 4; ++c) {
        const __bf16* bcol = tb + (size_t)((ct0 + c) * 16 + m) * F_;
        v16bf b0 = mk16(ld8(bcol + 8 * hi),      ld8(bcol + 16 + 8 * hi));
        v16bf b1 = mk16(ld8(bcol + 32 + 8 * hi), ld8(bcol + 48 + 8 * hi));
        acc[c] = __builtin_amdgcn_wmma_f32_16x16x32_bf16(
            false, a0, false, b0, (short)0, acc[c], false, false);
        acc[c] = __builtin_amdgcn_wmma_f32_16x16x32_bf16(
            false, a1, false, b1, (short)0, acc[c], false, false);
    }
    float* Ab = aout + (size_t)b * NN_;
    #pragma unroll
    for (int c = 0; c < 4; ++c) {
        #pragma unroll
        for (int r = 0; r < 8; ++r) {
            int rr = rowTile * 16 + r + hi * 8;
            Ab[(size_t)rr * N_ + (ct0 + c) * 16 + m] = acc[c][r];
        }
    }
}

// ---------------------------------------------------------------------------
// K6: zero histogram; K7: per-batch 1024-bin histogram of order-preserving keys
// ---------------------------------------------------------------------------
__global__ void k_zero(unsigned* __restrict__ p, int count) {
    int stride = gridDim.x * blockDim.x;
    for (int i = blockIdx.x * blockDim.x + threadIdx.x; i < count; i += stride) p[i] = 0u;
}

__global__ void k_hist(const float* __restrict__ a, unsigned* __restrict__ hist) {
    __shared__ unsigned sh[1024];
    int b = blockIdx.y;
    for (int i = threadIdx.x; i < 1024; i += blockDim.x) sh[i] = 0u;
    __syncthreads();
    const float* Ab = a + (size_t)b * NN_;
    int stride = gridDim.x * blockDim.x;
    for (int i = blockIdx.x * blockDim.x + threadIdx.x; i < NN_; i += stride) {
        unsigned u   = __float_as_uint(Ab[i]);
        unsigned key = (u & 0x80000000u) ? ~u : (u | 0x80000000u);
        atomicAdd(&sh[key >> 22], 1u);
    }
    __syncthreads();
    unsigned* gh = hist + b * 1024;
    for (int i = threadIdx.x; i < 1024; i += blockDim.x)
        if (sh[i]) atomicAdd(&gh[i], sh[i]);
}

// ---------------------------------------------------------------------------
// K8: reverse-scan histogram to find threshold bin for top-k
// ---------------------------------------------------------------------------
__global__ void k_scan(const unsigned* __restrict__ hist, const int* __restrict__ kvals,
                       int* __restrict__ thresh) {
    int b = blockIdx.x;
    if (threadIdx.x != 0) return;
    const unsigned* h = hist + b * 1024;
    unsigned k = (unsigned)kvals[b];
    unsigned cum = 0;
    int t = 0;
    for (int bin = 1023; bin >= 0; --bin) {
        cum += h[bin];
        if (cum >= k) { t = bin; break; }
    }
    thresh[b] = t;
}

// ---------------------------------------------------------------------------
// K9: mask = (a in top bins) | diagonal; non-temporal stores (never re-read)
// ---------------------------------------------------------------------------
__global__ void k_mask(const float* __restrict__ a, const int* __restrict__ thresh,
                       float* __restrict__ mask) {
    size_t total  = (size_t)B_ * NN_;
    size_t stride = (size_t)gridDim.x * blockDim.x;
    for (size_t i = (size_t)blockIdx.x * blockDim.x + threadIdx.x; i < total; i += stride) {
        size_t b = i / NN_;
        size_t r = i % NN_;
        int row = (int)(r / N_), colv = (int)(r % N_);
        unsigned u   = __float_as_uint(a[i]);
        unsigned key = (u & 0x80000000u) ? ~u : (u | 0x80000000u);
        int bin = (int)(key >> 22);
        float mv = ((bin >= thresh[b]) || (row == colv)) ? 1.0f : 0.0f;
        __builtin_nontemporal_store(mv, &mask[i]);
    }
}

// ---------------------------------------------------------------------------
extern "C" void kernel_launch(void* const* d_in, const int* in_sizes, int n_in,
                              void* d_out, int out_size, void* d_ws, size_t ws_size,
                              hipStream_t stream) {
    (void)in_sizes; (void)n_in; (void)out_size; (void)ws_size;
    const float* x_seq = (const float*)d_in[0];
    const float* prior = (const float*)d_in[1];
    const float* w1    = (const float*)d_in[2];
    const float* b1    = (const float*)d_in[3];
    const float* w2    = (const float*)d_in[4];
    const float* b2    = (const float*)d_in[5];
    const float* w3    = (const float*)d_in[6];
    const float* b3    = (const float*)d_in[7];
    const float* w4    = (const float*)d_in[8];
    const float* b4    = (const float*)d_in[9];
    const float* w_out = (const float*)d_in[10];
    const float* b_out = (const float*)d_in[11];

    char* ws = (char*)d_ws;
    __bf16*   xn     = (__bf16*)(ws + 0);               // (B,N,F)  bf16: 2 MB
    __bf16*   xt     = (__bf16*)(ws + 2097152);         // (B,F,N)  bf16: 2 MB
    __bf16*   pt     = (__bf16*)(ws + 4194304);         // (N,N)    bf16: 8 MB (P^T)
    __bf16*   tt     = (__bf16*)(ws + 12582912);        // (B,N,F)  bf16: 2 MB (T^T)
    float*    m2     = (float*)(ws + 14680064);         // B*L*F f32
    unsigned* hist   = (unsigned*)(ws + 14721024);      // B*1024 u32
    int*      kvals  = (int*)(ws + 14753792);           // B ints
    int*      thresh = (int*)(ws + 14753824);           // B ints

    float* mask_out = (float*)d_out;                    // (mask, a) concatenated
    float* a_out    = mask_out + (size_t)B_ * NN_;

    k_mean<<<B_ * L_, 64, 0, stream>>>(x_seq, m2);
    k_feats<<<B_, 256, 0, stream>>>(m2, w1, b1, w2, b2, w3, b3, w4, b4, w_out, b_out, kvals);
    k_normalize<<<B_ * N_, 32, 0, stream>>>(x_seq, xn, xt);
    k_transpose_bf16<<<dim3(64, 64), dim3(32, 8), 0, stream>>>(prior, pt);
    k_gemm1<<<dim3(128, B_), 32, 0, stream>>>(xt, pt, tt);
    k_gemm2<<<dim3(4, 128, B_), 256, 0, stream>>>(xn, tt, a_out);
    k_zero<<<32, 256, 0, stream>>>(hist, B_ * 1024);
    k_hist<<<dim3(256, B_), 256, 0, stream>>>(a_out, hist);
    k_scan<<<B_, 32, 0, stream>>>(hist, kvals, thresh);
    k_mask<<<4096, 256, 0, stream>>>(a_out, thresh, mask_out);
}
